// GRU_824633721297
// MI455X (gfx1250) — matrix operand order repaired and checked by reading, not compiled, not run on hardware
//
#include <hip/hip_runtime.h>

// ---------------------------------------------------------------------------
// Problem constants (match reference)
// ---------------------------------------------------------------------------
#define BB   256
#define TT   512
#define DIN  64
#define HH   512
#define NOUT 128
#define G3   (3 * HH)

typedef __bf16 bf16_t;
typedef __attribute__((ext_vector_type(16))) __bf16 v16bf;
typedef __attribute__((ext_vector_type(8)))  __bf16 v8bf;
typedef __attribute__((ext_vector_type(8)))  float  v8f;

// ---------------------------------------------------------------------------
// WMMA helpers (gfx1250 wave32, D = A(16x32 bf16) * B(32x16 bf16) + C(16x16 f32))
// ---------------------------------------------------------------------------
__device__ __forceinline__ v8f wmma_bf16(v16bf a, v16bf b, v8f c) {
    // (neg_a, A, neg_b, B, c_mod, C, reuse_a, reuse_b)
    return __builtin_amdgcn_wmma_f32_16x16x32_bf16(false, a, false, b,
                                                   (short)0, c, false, false);
}

// A operand, 16x32 bf16 tile, row-major source.
// Lane layout: lanes 0-15 -> M=lane, K = k0+[0..15]
//              lanes 16-31 -> M=lane-16, K = k0+[16..31]
// row_ptr must already point at A row (b0 + lane%16).
__device__ __forceinline__ v16bf load_a_tile(const bf16_t* row_ptr, int k0, int lane) {
    const int kb = k0 + ((lane >> 4) << 4);           // +0 or +16
    v8bf lo = *(const v8bf*)(row_ptr + kb);           // K = kb .. kb+7
    v8bf hi = *(const v8bf*)(row_ptr + kb + 8);       // K = kb+8 .. kb+15
    return __builtin_shufflevector(lo, hi, 0,1,2,3,4,5,6,7,8,9,10,11,12,13,14,15);
}

// B operand, 32x16 bf16 tile from row-major W (row = output channel, col = K).
// B[k][n] = W[grow0 + n][k]. Lane layout: lane n<16 -> N=n, K = k0+[0..15]
//                                         lane n>=16 -> N=n-16, K = k0+[16..31]
// i.e. ONE contiguous 32-byte load per lane.
__device__ __forceinline__ v16bf load_b_tile(const bf16_t* w, int ld, int grow0,
                                             int k0, int lane) {
    const int row = grow0 + (lane & 15);
    const int kb  = k0 + ((lane >> 4) << 4);          // +0 or +16
    return *(const v16bf*)(w + (size_t)row * ld + kb);
}

__device__ __forceinline__ float fast_sigmoid(float x) {
    return 1.0f / (1.0f + __expf(-x));
}
__device__ __forceinline__ float fast_tanh(float x) {
    return 2.0f / (1.0f + __expf(-2.0f * x)) - 1.0f;
}

// ---------------------------------------------------------------------------
// One 16x16 h-tile of a GRU step.
//   pre_r = x@Wih_r^T + h@Whh_r^T   (+ b_ih_r + b_hh_r)
//   pre_z = x@Wih_z^T + h@Whh_z^T   (+ b_ih_z + b_hh_z)
//   xn    = x@Wih_n^T               (+ b_ih_n)
//   hn    = h@Whh_n^T               (+ b_hh_n)   <- kept separate: r * hn
//   h' = (1-z)*tanh(xn + r*hn) + z*h
// ---------------------------------------------------------------------------
__device__ __forceinline__ void gru_tile(
    const bf16_t* __restrict__ xbase, int x_row_stride, int xK,
    const bf16_t* __restrict__ wih,
    const bf16_t* __restrict__ hprev_b, const float* __restrict__ hprev_f,
    const bf16_t* __restrict__ whh,
    const float* __restrict__ bih, const float* __restrict__ bhh,
    float* __restrict__ hout_f, bf16_t* __restrict__ hout_b,
    int b0, int j0, int lane)
{
    const int lm = lane & 15;
    v8f cr = {}, cz = {}, cxn = {}, chn = {};

    // Recurrent part: h_prev (K = HH)
    const bf16_t* Ah = hprev_b + (size_t)(b0 + lm) * HH;
    for (int k0 = 0; k0 < HH; k0 += 32) {
        v16bf a  = load_a_tile(Ah, k0, lane);
        v16bf br = load_b_tile(whh, HH, j0,          k0, lane);
        v16bf bz = load_b_tile(whh, HH, j0 + HH,     k0, lane);
        v16bf bn = load_b_tile(whh, HH, j0 + 2 * HH, k0, lane);
        cr  = wmma_bf16(a, br, cr);
        cz  = wmma_bf16(a, bz, cz);
        chn = wmma_bf16(a, bn, chn);
    }

    // Input-projection part: x_t (K = xK), folded into the step.
    const bf16_t* Ax = xbase + (size_t)(b0 + lm) * x_row_stride;
    for (int k0 = 0; k0 < xK; k0 += 32) {
        v16bf a  = load_a_tile(Ax, k0, lane);
        v16bf br = load_b_tile(wih, xK, j0,          k0, lane);
        v16bf bz = load_b_tile(wih, xK, j0 + HH,     k0, lane);
        v16bf bn = load_b_tile(wih, xK, j0 + 2 * HH, k0, lane);
        cr  = wmma_bf16(a, br, cr);
        cz  = wmma_bf16(a, bz, cz);
        cxn = wmma_bf16(a, bn, cxn);
    }

    // Epilogue: gate math. f32 C/D layout: VGPR r -> row (b0+r or b0+8+r), lane%16 -> col.
    const int   jc     = j0 + lm;
    const float bias_r = bih[jc]          + bhh[jc];
    const float bias_z = bih[HH + jc]     + bhh[HH + jc];
    const float bias_xn = bih[2 * HH + jc];
    const float bias_hn = bhh[2 * HH + jc];
    const int   mbase  = b0 + ((lane >> 4) << 3);

#pragma unroll
    for (int rr = 0; rr < 8; ++rr) {
        const int m = mbase + rr;
        const float rg = fast_sigmoid(cr[rr] + bias_r);
        const float zg = fast_sigmoid(cz[rr] + bias_z);
        const float ng = fast_tanh(cxn[rr] + bias_xn + rg * (chn[rr] + bias_hn));
        const float hp = hprev_f[(size_t)m * HH + jc];
        const float hn = (1.0f - zg) * ng + zg * hp;
        hout_f[(size_t)m * HH + jc] = hn;
        hout_b[(size_t)m * HH + jc] = (bf16_t)hn;
    }
}

// ---------------------------------------------------------------------------
// Pipelined stage kernel: stage s runs layer-0 step s and layer-1 step s-1.
// 1024 waves total: waves [0,512) layer 0, [512,1024) layer 1.
// ---------------------------------------------------------------------------
struct GruStageArgs {
    const bf16_t* xb;                       // (B,T,DIN) bf16
    const bf16_t* wih0; const bf16_t* whh0;
    const float*  bih0; const float*  bhh0;
    const bf16_t* wih1; const bf16_t* whh1;
    const float*  bih1; const float*  bhh1;
    bf16_t* hA_b[2]; float* hA_f[2];        // layer-0 hidden (double-buffered)
    bf16_t* hB_b[2]; float* hB_f[2];        // layer-1 hidden (double-buffered)
    int s;
};

__global__ __launch_bounds__(128) void gru_stage_kernel(GruStageArgs p) {
    const int lane = threadIdx.x & 31;
    const int wave = blockIdx.x * (blockDim.x >> 5) + (threadIdx.x >> 5);
    const int s    = p.s;

    if (wave < 512) {
        if (s >= TT) return;                               // layer 0 done
        const int tile = wave;
        const int b0 = (tile & 15) << 4;
        const int j0 = (tile >> 4) << 4;
        gru_tile(p.xb + (size_t)s * DIN, TT * DIN, DIN, p.wih0,
                 p.hA_b[s & 1], p.hA_f[s & 1], p.whh0, p.bih0, p.bhh0,
                 p.hA_f[(s + 1) & 1], p.hA_b[(s + 1) & 1], b0, j0, lane);
    } else {
        if (s < 1) return;                                 // layer 1 not started
        const int t1 = s - 1;
        const int tile = wave - 512;
        const int b0 = (tile & 15) << 4;
        const int j0 = (tile >> 4) << 4;
        // layer-1 "x" input is layer-0 output at t1, written at stage t1 into buf (t1+1)&1
        gru_tile(p.hA_b[(t1 + 1) & 1], HH, HH, p.wih1,
                 p.hB_b[t1 & 1], p.hB_f[t1 & 1], p.whh1, p.bih1, p.bhh1,
                 p.hB_f[(t1 + 1) & 1], p.hB_b[(t1 + 1) & 1], b0, j0, lane);
    }
}

// ---------------------------------------------------------------------------
// Final FC: out = h2_last @ w_fc^T + b_fc     (256x512 @ 512x128)
// ---------------------------------------------------------------------------
__global__ __launch_bounds__(128) void fc_kernel(const bf16_t* __restrict__ h,
                                                 const bf16_t* __restrict__ wfc,
                                                 const float*  __restrict__ bfc,
                                                 float* __restrict__ out) {
    const int lane = threadIdx.x & 31;
    const int wave = blockIdx.x * (blockDim.x >> 5) + (threadIdx.x >> 5);
    if (wave >= 128) return;
    const int b0 = (wave & 15) << 4;      // 16 row tiles
    const int o0 = (wave >> 4) << 4;      // 8 col tiles
    const int lm = lane & 15;

    v8f acc = {};
    const bf16_t* Ah = h + (size_t)(b0 + lm) * HH;
    for (int k0 = 0; k0 < HH; k0 += 32) {
        v16bf a = load_a_tile(Ah, k0, lane);
        v16bf b = load_b_tile(wfc, HH, o0, k0, lane);
        acc = wmma_bf16(a, b, acc);
    }
    const int oc = o0 + lm;
    const float bias = bfc[oc];
    const int mbase = b0 + ((lane >> 4) << 3);
#pragma unroll
    for (int rr = 0; rr < 8; ++rr)
        out[(size_t)(mbase + rr) * NOUT + oc] = acc[rr] + bias;
}

// ---------------------------------------------------------------------------
// Utility kernels
// ---------------------------------------------------------------------------
__global__ void cvt_f32_bf16_kernel(const float* __restrict__ src,
                                    bf16_t* __restrict__ dst, int n) {
    for (int i = blockIdx.x * blockDim.x + threadIdx.x; i < n;
         i += gridDim.x * blockDim.x)
        dst[i] = (bf16_t)src[i];
}

__global__ void zero_kernel(uint4* __restrict__ p, int n16) {
    for (int i = blockIdx.x * blockDim.x + threadIdx.x; i < n16;
         i += gridDim.x * blockDim.x)
        p[i] = make_uint4(0u, 0u, 0u, 0u);
}

// ---------------------------------------------------------------------------
// Host launch
// ---------------------------------------------------------------------------
extern "C" void kernel_launch(void* const* d_in, const int* in_sizes, int n_in,
                              void* d_out, int out_size, void* d_ws, size_t ws_size,
                              hipStream_t stream) {
    const float* x_f    = (const float*)d_in[0];
    const float* wih0_f = (const float*)d_in[1];
    const float* whh0_f = (const float*)d_in[2];
    const float* bih0   = (const float*)d_in[3];
    const float* bhh0   = (const float*)d_in[4];
    const float* wih1_f = (const float*)d_in[5];
    const float* whh1_f = (const float*)d_in[6];
    const float* bih1   = (const float*)d_in[7];
    const float* bhh1   = (const float*)d_in[8];
    const float* wfc_f  = (const float*)d_in[9];
    const float* bfc    = (const float*)d_in[10];
    float* out = (float*)d_out;

    // ---- workspace sub-allocation (256B aligned); total ~25 MB ----
    size_t off = 0;
    auto alloc = [&](size_t bytes) -> char* {
        char* p = (char*)d_ws + off;
        off += (bytes + 255) & ~(size_t)255;
        return p;
    };
    bf16_t* xb    = (bf16_t*)alloc((size_t)BB * TT * DIN * sizeof(bf16_t));
    bf16_t* wih0b = (bf16_t*)alloc((size_t)G3 * DIN * sizeof(bf16_t));
    bf16_t* whh0b = (bf16_t*)alloc((size_t)G3 * HH * sizeof(bf16_t));
    bf16_t* wih1b = (bf16_t*)alloc((size_t)G3 * HH * sizeof(bf16_t));
    bf16_t* whh1b = (bf16_t*)alloc((size_t)G3 * HH * sizeof(bf16_t));
    bf16_t* wfcb  = (bf16_t*)alloc((size_t)NOUT * HH * sizeof(bf16_t));

    char* state_base = (char*)d_ws + off;
    bf16_t* hA_b0 = (bf16_t*)alloc((size_t)BB * HH * sizeof(bf16_t));
    bf16_t* hA_b1 = (bf16_t*)alloc((size_t)BB * HH * sizeof(bf16_t));
    float*  hA_f0 = (float*) alloc((size_t)BB * HH * sizeof(float));
    float*  hA_f1 = (float*) alloc((size_t)BB * HH * sizeof(float));
    bf16_t* hB_b0 = (bf16_t*)alloc((size_t)BB * HH * sizeof(bf16_t));
    bf16_t* hB_b1 = (bf16_t*)alloc((size_t)BB * HH * sizeof(bf16_t));
    float*  hB_f0 = (float*) alloc((size_t)BB * HH * sizeof(float));
    float*  hB_f1 = (float*) alloc((size_t)BB * HH * sizeof(float));
    const size_t state_bytes = (size_t)(((char*)d_ws + off) - state_base);

    // ---- zero h states (buf0 is consumed at step 0; buf1 fully overwritten) ----
    zero_kernel<<<1024, 256, 0, stream>>>((uint4*)state_base, (int)(state_bytes / 16));

    // ---- convert inputs/weights to bf16 once ----
    cvt_f32_bf16_kernel<<<2048, 256, 0, stream>>>(x_f,    xb,    BB * TT * DIN);
    cvt_f32_bf16_kernel<<<256,  256, 0, stream>>>(wih0_f, wih0b, G3 * DIN);
    cvt_f32_bf16_kernel<<<1024, 256, 0, stream>>>(whh0_f, whh0b, G3 * HH);
    cvt_f32_bf16_kernel<<<1024, 256, 0, stream>>>(wih1_f, wih1b, G3 * HH);
    cvt_f32_bf16_kernel<<<1024, 256, 0, stream>>>(whh1_f, whh1b, G3 * HH);
    cvt_f32_bf16_kernel<<<256,  256, 0, stream>>>(wfc_f,  wfcb,  NOUT * HH);

    // ---- pipelined scan: 513 stages, layer0 step s + layer1 step s-1 ----
    GruStageArgs a;
    a.xb = xb;
    a.wih0 = wih0b; a.whh0 = whh0b; a.bih0 = bih0; a.bhh0 = bhh0;
    a.wih1 = wih1b; a.whh1 = whh1b; a.bih1 = bih1; a.bhh1 = bhh1;
    a.hA_b[0] = hA_b0; a.hA_b[1] = hA_b1;
    a.hA_f[0] = hA_f0; a.hA_f[1] = hA_f1;
    a.hB_b[0] = hB_b0; a.hB_b[1] = hB_b1;
    a.hB_f[0] = hB_f0; a.hB_f[1] = hB_f1;

    for (int s = 0; s <= TT; ++s) {
        a.s = s;
        gru_stage_kernel<<<256, 128, 0, stream>>>(a);
    }

    // ---- final FC: h2 last step (t1=511) lives in buf (511+1)&1 == 0 ----
    fc_kernel<<<32, 128, 0, stream>>>(hB_b0, wfcb, bfc, out);
}
